// Model_65025804861875
// MI455X (gfx1250) — matrix operand order repaired and checked by reading, not compiled, not run on hardware
//
#include <hip/hip_runtime.h>
#include <hip/hip_bf16.h>
#include <math.h>

typedef __attribute__((ext_vector_type(16))) __bf16 v16bf;
typedef __attribute__((ext_vector_type(8)))  float  v8f;

#define D128 128

// ---------------------------------------------------------------------------
// Small utility kernels
// ---------------------------------------------------------------------------
__global__ void degree_kernel(const int* __restrict__ src, float* __restrict__ deg, int E) {
    int i = blockIdx.x * blockDim.x + threadIdx.x;
    if (i < E) atomicAdd(&deg[src[i]], 1.0f);
}

__global__ void dinv_kernel(const float* __restrict__ deg, float* __restrict__ dinv, int N) {
    int i = blockIdx.x * blockDim.x + threadIdx.x;
    if (i < N) {
        float d = deg[i];
        dinv[i] = d > 0.0f ? rsqrtf(d) : 0.0f;
    }
}

__global__ void f32_to_bf16_kernel(const float* __restrict__ in, __bf16* __restrict__ out, int n) {
    int i = blockIdx.x * blockDim.x + threadIdx.x;
    if (i < n) out[i] = (__bf16)in[i];
}

// One wave (32 lanes) per edge; 4 floats per lane (D=128).
__global__ __launch_bounds__(256) void scatter_kernel(
    const int* __restrict__ src, const int* __restrict__ dst,
    const float* __restrict__ feats, const float* __restrict__ dinv,
    float* __restrict__ agg, int E)
{
    int t = blockIdx.x * blockDim.x + threadIdx.x;
    int e = t >> 5;
    if (e >= E) return;
    int lane = t & 31;
    int s = src[e], d = dst[e];
    float scl = dinv[s];
    const float4 v = ((const float4*)(feats + (size_t)s * D128))[lane];
    float* ap = agg + (size_t)d * D128 + lane * 4;
    atomicAdd(ap + 0, v.x * scl);
    atomicAdd(ap + 1, v.y * scl);
    atomicAdd(ap + 2, v.z * scl);
    atomicAdd(ap + 3, v.w * scl);
}

__global__ void pred_kernel(const float* __restrict__ pv, const int* __restrict__ eli,
                            float* __restrict__ out, int K) {
    int i = blockIdx.x * blockDim.x + threadIdx.x;
    if (i < K) out[i] = pv[eli[i]] * pv[eli[K + i]];
}

// ---------------------------------------------------------------------------
// Generic GEMM: C[M,N] = act( (rowScale?A*rowScale:A)[M,K] @ W[N,K]^T + bias + Cin )
// bf16 WMMA, f32 accumulate. One 16x16 tile / wave. Optional bf16 output.
// ---------------------------------------------------------------------------
__global__ __launch_bounds__(128) void gemm_bf16_wmma(
    const float* __restrict__ A, const float* __restrict__ rowScale,
    const __bf16* __restrict__ W,           // N x K, row-major, bf16
    const float* __restrict__ bias, const float* __restrict__ Cin,
    float* __restrict__ C, __bf16* __restrict__ Cbf,
    int M, int N, int K, int act)
{
    const int lane = threadIdx.x & 31;
    const int wave = threadIdx.x >> 5;
    const int tilesN = N >> 4;
    const int tile = blockIdx.x * 4 + wave;
    const int tm = tile / tilesN, tn = tile % tilesN;
    if (tm * 16 >= M) return;
    const int half = lane >> 4;
    const int l16  = lane & 15;
    const int rowA = tm * 16 + l16;       // A-layout: lane's matrix row
    const int colC = tn * 16 + l16;       // B/C layout: lane's column
    const float scale = rowScale ? rowScale[rowA] : 1.0f;

    v8f c;
    const float base = bias ? bias[colC] : 0.0f;
#pragma unroll
    for (int r = 0; r < 8; ++r) {
        int row = tm * 16 + r + half * 8;
        c[r] = (Cin ? Cin[(size_t)row * N + colC] : 0.0f) + base;
    }

    for (int kk = 0; kk < K; kk += 32) {
        if (kk + 32 < K)
            __builtin_prefetch(A + (size_t)rowA * K + kk + 32, 0, 0);
        v16bf a;
#pragma unroll
        for (int i = 0; i < 16; ++i) {
            int k = kk + i + half * 16;   // contiguous k order (matches B)
            a[i] = (__bf16)(A[(size_t)rowA * K + k] * scale);
        }
        // B layout: lane = column, 16 contiguous bf16 along K  -> b128 loads
        const __bf16* wp = W + (size_t)colC * K + kk + half * 16;
        v16bf b = *(const v16bf*)wp;
        c = __builtin_amdgcn_wmma_f32_16x16x32_bf16(false, a, false, b, (short)0, c, false, false);
    }

#pragma unroll
    for (int r = 0; r < 8; ++r) {
        int row = tm * 16 + r + half * 8;
        float v = c[r];
        if (act == 1) v = fmaxf(v, 0.0f);
        if (Cbf) Cbf[(size_t)row * N + colC] = (__bf16)v;
        else     C[(size_t)row * N + colC]   = v;
    }
}

// ---------------------------------------------------------------------------
// Flash-style cross attention, 4 waves / 4 m-tiles per block (one head).
// K tiles: async DMA -> LDS (global_load_async_to_lds_b128), double-buffered.
// V tiles: staged transposed into LDS so the PV B-operand is contiguous.
// scores tile = one 16x16x32 bf16 WMMA (head dim == 32); online softmax.
// ---------------------------------------------------------------------------
__device__ __forceinline__ void wait_async_le1(int pending) {
    if (pending) asm volatile("s_wait_asynccnt 0x1" ::: "memory");
    else         asm volatile("s_wait_asynccnt 0x0" ::: "memory");
}

__global__ __launch_bounds__(128) void attn_kernel(
    const float* __restrict__ q, const __bf16* __restrict__ kbf,
    const __bf16* __restrict__ vbf, float* __restrict__ o, int S)
{
    __shared__ __attribute__((aligned(32))) __bf16 ldsK[2][16 * 32];  // [seq][dim]
    __shared__ __attribute__((aligned(32))) __bf16 ldsVT[2][32 * 16]; // [dim][seq]
    __shared__ float pT[4][16 * 16];
    const int h    = blockIdx.y;
    const int tid  = threadIdx.x;
    const int lane = tid & 31;
    const int wave = tid >> 5;
    const int half = lane >> 4, l16 = lane & 15;
    const int tm   = blockIdx.x * 4 + wave;
    const int HD = 32;
    const float scq = 0.17677669529663687f;   // 1/sqrt(32)

    // Q tile, A layout, pre-scaled
    v16bf qa;
    const int rowA = tm * 16 + l16;
#pragma unroll
    for (int i = 0; i < 16; ++i) {
        int k = i + half * 16;
        qa[i] = (__bf16)(q[(size_t)rowA * D128 + h * HD + k] * scq);
    }

    float rm[8], rs[8];
    v8f o0 = {}, o1 = {};
#pragma unroll
    for (int r = 0; r < 8; ++r) { rm[r] = -1e30f; rs[r] = 0.0f; }
    float* myP = &pT[wave][0];

    // cooperative tile staging: waves 0-1 DMA K, waves 2-3 transpose V
    auto stage = [&](int t) {
        int buf = t & 1;
        int s0  = t * 16;
        if (tid < 64) {
            int row = tid >> 2, chunk = (tid & 3) * 8;           // 8 bf16 = 16B each
            const __bf16* sp = kbf + (size_t)(s0 + row) * D128 + h * HD + chunk;
            unsigned ldsoff = (unsigned)(size_t)&ldsK[buf][row * 32 + chunk];
            asm volatile("global_load_async_to_lds_b128 %0, %1, off"
                         :: "v"(ldsoff), "v"(sp) : "memory");
        } else {
            int u = tid - 64;
            int seq = u >> 2, dgrp = (u & 3) * 8;
            const __bf16* vp = vbf + (size_t)(s0 + seq) * D128 + h * HD + dgrp;
            __bf16 tmp[8];
            *(uint4*)tmp = *(const uint4*)vp;
#pragma unroll
            for (int j = 0; j < 8; ++j)
                ldsVT[buf][(dgrp + j) * 16 + seq] = tmp[j];
        }
    };

    const int ST = S / 16;
    stage(0);
    for (int t = 0; t < ST; ++t) {
        const int cur = t & 1;
        const int has_next = (t + 1 < ST);
        if (has_next) stage(t + 1);
        wait_async_le1(has_next);    // our own async op for tile t has landed
        __syncthreads();             // all waves' staging of buffer `cur` visible

        // ---- scores: B[k][n] = K[s0+n][k], contiguous in ldsK ----
        v16bf kb = *(const v16bf*)&ldsK[cur][l16 * 32 + half * 16];
        v8f z = {};
        v8f scs = __builtin_amdgcn_wmma_f32_16x16x32_bf16(false, qa, false, kb, (short)0, z, false, false);

        // ---- online softmax (half-wave reductions match C layout) ----
        float pr[8];
#pragma unroll
        for (int r = 0; r < 8; ++r) {
            float mt = scs[r];
#pragma unroll
            for (int m = 1; m < 16; m <<= 1) mt = fmaxf(mt, __shfl_xor(mt, m, 16));
            float mnew = fmaxf(rm[r], mt);
            float corr = __expf(rm[r] - mnew);
            float p    = __expf(scs[r] - mnew);
            float ps = p;
#pragma unroll
            for (int m = 1; m < 16; m <<= 1) ps += __shfl_xor(ps, m, 16);
            rs[r] = rs[r] * corr + ps;
            rm[r] = mnew;
            o0[r] *= corr; o1[r] *= corr;
            pr[r] = p;
        }

        // ---- P: C layout -> A layout through per-wave LDS (wave-local fence) ----
#pragma unroll
        for (int r = 0; r < 8; ++r)
            myP[(r + half * 8) * 16 + l16] = pr[r];
        asm volatile("s_wait_dscnt 0x0" ::: "memory");
        v16bf pa;
#pragma unroll
        for (int i = 0; i < 16; ++i) {
            int k = i + half * 16;
            pa[i] = (k < 16) ? (__bf16)myP[l16 * 16 + k] : (__bf16)0.0f;
        }

        // ---- P @ V: B from transposed V in LDS, K zero-padded 16->32 ----
        v16bf zv = {};
        v16bf vb0 = half ? zv : *(const v16bf*)&ldsVT[cur][l16 * 16];
        v16bf vb1 = half ? zv : *(const v16bf*)&ldsVT[cur][(16 + l16) * 16];
        o0 = __builtin_amdgcn_wmma_f32_16x16x32_bf16(false, pa, false, vb0, (short)0, o0, false, false);
        o1 = __builtin_amdgcn_wmma_f32_16x16x32_bf16(false, pa, false, vb1, (short)0, o1, false, false);

        __syncthreads();             // buffer `cur` free for reuse
    }

#pragma unroll
    for (int r = 0; r < 8; ++r) {
        float inv = rs[r] > 0.0f ? 1.0f / rs[r] : 0.0f;
        int row = tm * 16 + r + half * 8;
        o[(size_t)row * D128 + h * HD + l16]      = o0[r] * inv;
        o[(size_t)row * D128 + h * HD + 16 + l16] = o1[r] * inv;
    }
}

// ---------------------------------------------------------------------------
// Row-wise kernels (128 threads = 4 waves per row)
// ---------------------------------------------------------------------------
__global__ __launch_bounds__(128) void layernorm_kernel(
    float* __restrict__ x, const float* __restrict__ g, const float* __restrict__ b)
{
    __shared__ float red[4];
    const int row = blockIdx.x, tid = threadIdx.x;
    float v = x[(size_t)row * D128 + tid];
    float s = v;
#pragma unroll
    for (int m = 1; m < 32; m <<= 1) s += __shfl_xor(s, m, 32);
    if ((tid & 31) == 0) red[tid >> 5] = s;
    __syncthreads();
    float mu = (red[0] + red[1] + red[2] + red[3]) * (1.0f / D128);
    float d = v - mu;
    float s2 = d * d;
#pragma unroll
    for (int m = 1; m < 32; m <<= 1) s2 += __shfl_xor(s2, m, 32);
    __syncthreads();
    if ((tid & 31) == 0) red[tid >> 5] = s2;
    __syncthreads();
    float var = (red[0] + red[1] + red[2] + red[3]) * (1.0f / D128);
    x[(size_t)row * D128 + tid] = d * rsqrtf(var + 1e-5f) * g[tid] + b[tid];
}

__global__ __launch_bounds__(128) void l2norm_kernel(
    const float* __restrict__ x, float* __restrict__ out)
{
    __shared__ float red[4];
    const int row = blockIdx.x, tid = threadIdx.x;
    float v = x[(size_t)row * D128 + tid];
    float s = v * v;
#pragma unroll
    for (int m = 1; m < 32; m <<= 1) s += __shfl_xor(s, m, 32);
    if ((tid & 31) == 0) red[tid >> 5] = s;
    __syncthreads();
    float nrm = sqrtf(red[0] + red[1] + red[2] + red[3]);
    out[(size_t)row * D128 + tid] = v / fmaxf(nrm, 1e-12f);
}

__global__ __launch_bounds__(128) void dot_sigmoid_kernel(
    const float* __restrict__ p1, const float* __restrict__ w2, float* __restrict__ pv)
{
    __shared__ float red[4];
    const int row = blockIdx.x, tid = threadIdx.x;
    float s = p1[(size_t)row * D128 + tid] * w2[tid];
#pragma unroll
    for (int m = 1; m < 32; m <<= 1) s += __shfl_xor(s, m, 32);
    if ((tid & 31) == 0) red[tid >> 5] = s;
    __syncthreads();
    if (tid == 0) {
        float t = red[0] + red[1] + red[2] + red[3];
        pv[row] = 1.0f / (1.0f + __expf(-t));
    }
}

// ---------------------------------------------------------------------------
// Host orchestration
// ---------------------------------------------------------------------------
static inline void gemm(const float* A, const float* rowScale, const __bf16* W,
                        const float* bias, const float* Cin, float* C, __bf16* Cbf,
                        int M, int N, int K, int act, hipStream_t stream) {
    int tiles = (M / 16) * (N / 16);
    gemm_bf16_wmma<<<tiles / 4, 128, 0, stream>>>(A, rowScale, W, bias, Cin, C, Cbf, M, N, K, act);
}

extern "C" void kernel_launch(void* const* d_in, const int* in_sizes, int n_in,
                              void* d_out, int out_size, void* d_ws, size_t ws_size,
                              hipStream_t stream) {
    (void)n_in; (void)out_size; (void)ws_size;
    const float* x     = (const float*)d_in[0];
    const float* seq   = (const float*)d_in[1];
    const int*   src   = (const int*)d_in[2];
    const int*   dst   = (const int*)d_in[3];
    const int*   eli   = (const int*)d_in[4];
    const float* skw0  = (const float*)d_in[5];
    const float* skb0  = (const float*)d_in[6];
    const float* msw0  = (const float*)d_in[7];
    const float* msb0  = (const float*)d_in[8];
    const float* skw1  = (const float*)d_in[9];
    const float* skb1  = (const float*)d_in[10];
    const float* msw1  = (const float*)d_in[11];
    const float* msb1  = (const float*)d_in[12];
    const float* ipw   = (const float*)d_in[13];
    const float* ipb   = (const float*)d_in[14];
    const float* opw   = (const float*)d_in[15];
    const float* opb   = (const float*)d_in[16];
    const float* lng   = (const float*)d_in[17];
    const float* lnb   = (const float*)d_in[18];
    const float* w1    = (const float*)d_in[19];
    const float* w2    = (const float*)d_in[20];

    const int N = in_sizes[0] / D128;
    const int S = in_sizes[1] / D128;
    const int E = in_sizes[2];
    const int K = in_sizes[4] / 2;
    const int DD = D128 * D128;

    // workspace carve (floats, 64-float aligned)
    float* wsf = (float*)d_ws;
    size_t off = 0;
    auto alloc = [&](size_t nfl) { float* p = wsf + off; off += (nfl + 63) & ~(size_t)63; return p; };
    float* deg  = alloc(N);
    float* dinv = alloc(N);
    float* agg  = alloc((size_t)N * D128);
    float* xa   = alloc((size_t)N * D128);
    float* xb   = alloc((size_t)N * D128);
    float* qb   = alloc((size_t)N * D128);
    float* ao   = alloc((size_t)N * D128);
    float* xc   = alloc((size_t)N * D128);
    float* p1   = alloc((size_t)N * D128);
    float* pv   = alloc(N);
    __bf16* kbf = (__bf16*)alloc(((size_t)S * D128 + 1) / 2);
    __bf16* vbf = (__bf16*)alloc(((size_t)S * D128 + 1) / 2);
    __bf16* bfbase = (__bf16*)alloc((9 * (size_t)DD + 1) / 2);
    __bf16* msw0b = bfbase + 0 * DD;
    __bf16* skw0b = bfbase + 1 * DD;
    __bf16* msw1b = bfbase + 2 * DD;
    __bf16* skw1b = bfbase + 3 * DD;
    __bf16* ipwb  = bfbase + 4 * DD;    // 3*DD: wq, wk, wv
    __bf16* opwb  = bfbase + 7 * DD;
    __bf16* w1b   = bfbase + 8 * DD;

    float* pred     = (float*)d_out;
    float* node_emb = (float*)d_out + K;

    // degree / dinv (shared by both GCN layers)
    hipMemsetAsync(deg, 0, (size_t)N * sizeof(float), stream);
    degree_kernel<<<(E + 255) / 256, 256, 0, stream>>>(src, deg, E);
    dinv_kernel<<<(N + 255) / 256, 256, 0, stream>>>(deg, dinv, N);

    // weight down-conversion (once; all weights stay L2-resident)
    f32_to_bf16_kernel<<<(DD + 255) / 256, 256, 0, stream>>>(msw0, msw0b, DD);
    f32_to_bf16_kernel<<<(DD + 255) / 256, 256, 0, stream>>>(skw0, skw0b, DD);
    f32_to_bf16_kernel<<<(DD + 255) / 256, 256, 0, stream>>>(msw1, msw1b, DD);
    f32_to_bf16_kernel<<<(DD + 255) / 256, 256, 0, stream>>>(skw1, skw1b, DD);
    f32_to_bf16_kernel<<<(3 * DD + 255) / 256, 256, 0, stream>>>(ipw, ipwb, 3 * DD);
    f32_to_bf16_kernel<<<(DD + 255) / 256, 256, 0, stream>>>(opw, opwb, DD);
    f32_to_bf16_kernel<<<(DD + 255) / 256, 256, 0, stream>>>(w1, w1b, DD);

    // ---- GCN layer 0 ----
    hipMemsetAsync(agg, 0, (size_t)N * D128 * sizeof(float), stream);
    scatter_kernel<<<(E * 32 + 255) / 256, 256, 0, stream>>>(src, dst, x, dinv, agg, E);
    gemm(agg, dinv, msw0b, msb0, nullptr, xa, nullptr, N, D128, D128, 0, stream);
    gemm(x, nullptr, skw0b, skb0, xa, xa, nullptr, N, D128, D128, 0, stream);

    // ---- GCN layer 1 ----
    hipMemsetAsync(agg, 0, (size_t)N * D128 * sizeof(float), stream);
    scatter_kernel<<<(E * 32 + 255) / 256, 256, 0, stream>>>(src, dst, xa, dinv, agg, E);
    gemm(agg, dinv, msw1b, msb1, nullptr, xb, nullptr, N, D128, D128, 0, stream);
    gemm(xa, nullptr, skw1b, skb1, xb, xb, nullptr, N, D128, D128, 0, stream);

    // ---- cross attention (K/V projections write bf16 directly) ----
    gemm(xb,  nullptr, ipwb + 0 * DD, ipb + 0,        nullptr, qb, nullptr, N, D128, D128, 0, stream);
    gemm(seq, nullptr, ipwb + 1 * DD, ipb + D128,     nullptr, nullptr, kbf, S, D128, D128, 0, stream);
    gemm(seq, nullptr, ipwb + 2 * DD, ipb + 2 * D128, nullptr, nullptr, vbf, S, D128, D128, 0, stream);
    attn_kernel<<<dim3(N / 64, 4), 128, 0, stream>>>(qb, kbf, vbf, ao, S);
    gemm(ao, nullptr, opwb, opb, xb /*residual*/, xc, nullptr, N, D128, D128, 0, stream);
    layernorm_kernel<<<N, 128, 0, stream>>>(xc, lng, lnb);

    // ---- head ----
    l2norm_kernel<<<N, 128, 0, stream>>>(xc, node_emb);
    gemm(node_emb, nullptr, w1b, nullptr, nullptr, p1, nullptr, N, D128, D128, 1 /*relu*/, stream);
    dot_sigmoid_kernel<<<N, 128, 0, stream>>>(p1, w2, pv);
    pred_kernel<<<(K + 255) / 256, 256, 0, stream>>>(pv, eli, pred, K);
}
